// ContractLevelAttention_81587198755038
// MI455X (gfx1250) — matrix operand order repaired and checked by reading, not compiled, not run on hardware
//
#include <hip/hip_runtime.h>
#include <hip/hip_bf16.h>

#define D_IN  256
#define H_DIM 128

typedef __bf16 bf16_t;
typedef bf16_t v16bf __attribute__((ext_vector_type(16)));
typedef float  v8f   __attribute__((ext_vector_type(8)));

// round-to-nearest-even f32 -> bf16, pack two into a dword (lo = first)
__device__ __forceinline__ unsigned int packbf2(float a, float b) {
  unsigned int ua = __builtin_bit_cast(unsigned int, a);
  unsigned int ub = __builtin_bit_cast(unsigned int, b);
  ua += 0x7FFFu + ((ua >> 16) & 1u);
  ub += 0x7FFFu + ((ub >> 16) & 1u);
  return (ua >> 16) | (ub & 0xFFFF0000u);
}

// ---------------------------------------------------------------------------
// Kernel 1: scores[m] = tanh(x[m,:] @ W1 + b1) @ W2 + b2  via bf16 WMMA
// block = 256 threads (8 waves), each block does 128 rows of x.
// ---------------------------------------------------------------------------
__global__ __launch_bounds__(256)
void scores_kernel(const float* __restrict__ x,
                   const float* __restrict__ W1,
                   const float* __restrict__ b1,
                   const float* __restrict__ W2,
                   const float* __restrict__ b2,
                   float* __restrict__ scores)
{
  // W1 pre-swizzled into per-(kstep, ntile) bf16 B fragments:
  // fragment(ks,n): 32 lanes x 8 dwords (16 bf16) contiguous -> 1 LDS b128x2 per lane
  __shared__ __align__(32) unsigned int w1f[16384];   // 64 KB

  const int tid = threadIdx.x;
  for (int f = tid; f < 16384; f += 256) {
    int pairIdx = f & 7;            // which dword within the lane's fragment
    int lane    = (f >> 3) & 31;
    int n       = (f >> 8) & 7;     // N tile
    int ks      = f >> 11;          // K step (32 per step)
    // B fragment (32x16), this code's contiguous k order: lanes 0-15 hold K=0..15, lanes 16-31 K=16..31,
    // dword j holds K = 2j, 2j+1 (within the half)
    int col = n * 16 + (lane & 15);
    int row = ks * 32 + ((lane & 16) ? 16 : 0) + 2 * pairIdx;
    w1f[f] = packbf2(W1[row * H_DIM + col], W1[(row + 1) * H_DIM + col]);
  }
  __syncthreads();

  const int lane = tid & 31;
  const int wave = tid >> 5;
  const int m0   = blockIdx.x * 128 + wave * 16;   // this wave's 16 rows
  const int lm   = lane & 15;
  const float* xrow = x + (size_t)(m0 + lm) * D_IN;

  float b1v[8], w2v[8];
#pragma unroll
  for (int n = 0; n < 8; ++n) {
    b1v[n] = b1[n * 16 + lm];
    w2v[n] = W2[n * 16 + lm];
  }

  v8f zero = {};
  v8f acc[8];
#pragma unroll
  for (int n = 0; n < 8; ++n) acc[n] = zero;

  // A fragment (16x32), contiguous k order matching B: lanes 0-15 -> K 0..15; lanes 16-31 -> K 16..31
  const int khalf = (lane & 16) ? 16 : 0;
#pragma unroll
  for (int ks = 0; ks < 8; ++ks) {
    const int kb = ks * 32 + khalf;
    float4 f0 = *(const float4*)(xrow + kb);
    float4 f1 = *(const float4*)(xrow + kb + 4);
    float4 f2 = *(const float4*)(xrow + kb + 8);
    float4 f3 = *(const float4*)(xrow + kb + 12);
    union { unsigned int u[8]; v16bf v; } A;
    A.u[0] = packbf2(f0.x, f0.y);  A.u[1] = packbf2(f0.z, f0.w);
    A.u[2] = packbf2(f1.x, f1.y);  A.u[3] = packbf2(f1.z, f1.w);
    A.u[4] = packbf2(f2.x, f2.y);  A.u[5] = packbf2(f2.z, f2.w);
    A.u[6] = packbf2(f3.x, f3.y);  A.u[7] = packbf2(f3.z, f3.w);
#pragma unroll
    for (int n = 0; n < 8; ++n) {
      const v16bf* bp = (const v16bf*)&w1f[((ks * 8 + n) * 32 + lane) * 8];
      acc[n] = __builtin_amdgcn_wmma_f32_16x16x32_bf16(
          false, A.v, false, *bp, (short)0, acc[n], false, false);
    }
  }

  // C/D layout: dword r, lanes 0-15 -> row m0+r, col n*16+lane;
  //             lanes 16-31 -> row m0+8+r, col n*16+(lane-16)
  float p[8];
#pragma unroll
  for (int r = 0; r < 8; ++r) {
    float s = 0.0f;
#pragma unroll
    for (int n = 0; n < 8; ++n)
      s += tanhf(acc[n][r] + b1v[n]) * w2v[n];
    p[r] = s;
  }
  // reduce the 16 columns held across each 16-lane half-group
#pragma unroll
  for (int off = 1; off < 16; off <<= 1) {
#pragma unroll
    for (int r = 0; r < 8; ++r)
      p[r] += __shfl_xor(p[r], off, 32);
  }
  const float bias2 = b2[0];
  if (lm == 0) {
    const int rowbase = m0 + ((lane & 16) ? 8 : 0);
#pragma unroll
    for (int r = 0; r < 8; ++r)
      scores[rowbase + r] = p[r] + bias2;
  }
}

// ---------------------------------------------------------------------------
// Kernel 2: per-segment numerically stable softmax over scores (in place)
// ---------------------------------------------------------------------------
__device__ __forceinline__ int lower_bound_ll(const long long* __restrict__ a,
                                              int n, long long key) {
  int lo = 0, hi = n;
  while (lo < hi) {
    int mid = (lo + hi) >> 1;
    if (a[mid] < key) lo = mid + 1; else hi = mid;
  }
  return lo;
}

__global__ __launch_bounds__(256)
void seg_softmax_kernel(float* __restrict__ sw,               // scores in, weights out
                        const long long* __restrict__ batch,
                        int n_nodes)
{
  const int b   = blockIdx.x;
  const int tid = threadIdx.x;
  const int start = lower_bound_ll(batch, n_nodes, (long long)b);
  const int end   = lower_bound_ll(batch, n_nodes, (long long)b + 1);
  if (start >= end) return;

  __shared__ float red[8];
  const int lane = tid & 31, wave = tid >> 5;

  // pass 1: segment max
  float m = -3.402823466e38f;
  for (int i = start + tid; i < end; i += 256) m = fmaxf(m, sw[i]);
  for (int off = 16; off; off >>= 1) m = fmaxf(m, __shfl_xor(m, off, 32));
  if (lane == 0) red[wave] = m;
  __syncthreads();
  m = red[0];
#pragma unroll
  for (int j = 1; j < 8; ++j) m = fmaxf(m, red[j]);
  __syncthreads();

  // pass 2: sum of exp
  float s = 0.0f;
  for (int i = start + tid; i < end; i += 256) s += expf(sw[i] - m);
  for (int off = 16; off; off >>= 1) s += __shfl_xor(s, off, 32);
  if (lane == 0) red[wave] = s;
  __syncthreads();
  float denom = 0.0f;
#pragma unroll
  for (int j = 0; j < 8; ++j) denom += red[j];
  const float inv = 1.0f / denom;

  // pass 3: normalized weights, in place
  for (int i = start + tid; i < end; i += 256)
    sw[i] = expf(sw[i] - m) * inv;
}

// ---------------------------------------------------------------------------
// Kernel 3: pooled[b,:] = sum_i in seg(b) x[i,:] * w[i]   (coalesced, no atomics)
// ---------------------------------------------------------------------------
__global__ __launch_bounds__(256)
void pool_kernel(const float* __restrict__ x,
                 const float* __restrict__ w,
                 const long long* __restrict__ batch,
                 float* __restrict__ out,
                 int n_nodes)
{
  const int b = blockIdx.x;
  const int t = threadIdx.x;                 // column 0..255
  const int start = lower_bound_ll(batch, n_nodes, (long long)b);
  const int end   = lower_bound_ll(batch, n_nodes, (long long)b + 1);
  float acc = 0.0f;
  int i = start;
  for (; i + 3 < end; i += 4) {
    float w0 = w[i], w1 = w[i + 1], w2 = w[i + 2], w3 = w[i + 3];
    acc += x[(size_t)(i    ) * D_IN + t] * w0;
    acc += x[(size_t)(i + 1) * D_IN + t] * w1;
    acc += x[(size_t)(i + 2) * D_IN + t] * w2;
    acc += x[(size_t)(i + 3) * D_IN + t] * w3;
  }
  for (; i < end; ++i) acc += x[(size_t)i * D_IN + t] * w[i];
  out[(size_t)b * D_IN + t] = acc;           // empty segments -> zeros
}

// ---------------------------------------------------------------------------
extern "C" void kernel_launch(void* const* d_in, const int* in_sizes, int n_in,
                              void* d_out, int out_size, void* d_ws, size_t ws_size,
                              hipStream_t stream) {
  const float*     x     = (const float*)d_in[0];
  const long long* batch = (const long long*)d_in[1];
  const float*     W1    = (const float*)d_in[2];
  const float*     b1    = (const float*)d_in[3];
  const float*     W2    = (const float*)d_in[4];
  const float*     b2    = (const float*)d_in[5];
  float* out = (float*)d_out;

  const int N = in_sizes[0] / D_IN;     // 524288
  const int B = out_size / D_IN;        // 2048
  float* sw = (float*)d_ws;             // N floats: scores -> softmax weights

  scores_kernel<<<N / 128, 256, 0, stream>>>(x, W1, b1, W2, b2, sw);
  seg_softmax_kernel<<<B, 256, 0, stream>>>(sw, batch, N);
  pool_kernel<<<B, 256, 0, stream>>>(x, sw, batch, out, N);
}
